// NestedBlock_58875411693983
// MI455X (gfx1250) — matrix operand order repaired and checked by reading, not compiled, not run on hardware
//
#include <hip/hip_runtime.h>

// ---------------- types / WMMA helpers (gfx1250, wave32) ----------------
typedef __attribute__((ext_vector_type(16))) __bf16 v16bf;
typedef __attribute__((ext_vector_type(8)))  float  v8f;
typedef __attribute__((ext_vector_type(4)))  unsigned v4u;
typedef __attribute__((ext_vector_type(8)))  int      v8i;
typedef __attribute__((ext_vector_type(4)))  int      v4i;

#define BB   4
#define NN   2048
#define DD   512
#define HH   8
#define HDIM 64
#define MTOT (BB*NN)      /* 8192 tokens */
#define D3   1536
#define RD   2048

union FragU { v16bf v; unsigned u[8]; };

__device__ __forceinline__ v8f wmma_bf16(v16bf a, v16bf b, v8f c) {
  return __builtin_amdgcn_wmma_f32_16x16x32_bf16(false, a, false, b, (short)0, c, false, false);
}

// A fragment: 16x32 bf16, row = lane&15 (both halves), K split per ISA layout.
__device__ __forceinline__ v16bf a_frag32(const unsigned* row, int hs) {
  FragU f; const int b0 = hs ? 4 : 0;
#pragma unroll
  for (int v = 0; v < 8; ++v) f.u[v] = row[(v < 4 ? 0 : 8) + b0 + (v & 3)];
  return f.v;
}
// B fragment: 32x16 bf16, col = lane&15; K split per ISA layout (same as A).
__device__ __forceinline__ v16bf b_frag32(const unsigned* row, int hs) {
  FragU f; const int b0 = hs ? 4 : 0;
#pragma unroll
  for (int v = 0; v < 8; ++v) f.u[v] = row[(v < 4 ? 0 : 8) + b0 + (v & 3)];
  return f.v;
}

// DPP16 ROW_ROR rotations: VALU-speed reductions confined to each 16-lane row.
template <int CTRL>
__device__ __forceinline__ float dpp_rot(float x) {
  int i = __float_as_int(x);
  int r = __builtin_amdgcn_update_dpp(i, i, CTRL, 0xF, 0xF, false);
  return __int_as_float(r);
}
__device__ __forceinline__ float rmax16(float x) {
  x = fmaxf(x, dpp_rot<0x128>(x));   // row_ror:8
  x = fmaxf(x, dpp_rot<0x124>(x));   // row_ror:4
  x = fmaxf(x, dpp_rot<0x122>(x));   // row_ror:2
  x = fmaxf(x, dpp_rot<0x121>(x));   // row_ror:1
  return x;
}
__device__ __forceinline__ float rsum16(float x) {
  x += dpp_rot<0x128>(x);
  x += dpp_rot<0x124>(x);
  x += dpp_rot<0x122>(x);
  x += dpp_rot<0x121>(x);
  return x;
}

// ---------------- Tensor Data Mover (TDM) staging ----------------
// Builds a D# per CDNA5 ISA 8.3/8.4 and issues TENSOR_LOAD_TO_LDS.
// 2D tile: x = t0 contiguous elems, y = t1 rows with stride s0 (elements).
// 3D tile (t2>0): z dim with stride s1 — with t0==1 this lands the tile
// transposed in LDS (x->y->z fill order).
__device__ __forceinline__ void tdm_load(unsigned lds_off, const void* gptr,
                                         unsigned td0, unsigned td1, unsigned td2,
                                         unsigned t0, unsigned t1, unsigned t2,
                                         unsigned long long s0, unsigned long long s1) {
  unsigned long long ga = (unsigned long long)(uintptr_t)gptr;
  v4u g0 = {1u,                                   // count=1, user desc
            lds_off,                              // lds_addr (bytes)
            (unsigned)ga,                         // global_addr[31:0]
            (unsigned)((ga >> 32) & 0x01FFFFFFu) | (2u << 30)};  // addr[56:32] | type=2
  v8i g1 = {(int)(1u << 16),                                     // data_size=1 -> 2 bytes
            (int)((td0 & 0xFFFFu) << 16),                        // tensor_dim0 lo
            (int)((td0 >> 16) | ((td1 & 0xFFFFu) << 16)),        // td0 hi | td1 lo
            (int)((td1 >> 16) | ((t0 & 0xFFFFu) << 16)),         // td1 hi | tile_dim0
            (int)((t1 & 0xFFFFu) | ((t2 & 0xFFFFu) << 16)),      // tile_dim1 | tile_dim2
            (int)(unsigned)(s0 & 0xFFFFFFFFull),                 // dim0_stride[31:0]
            (int)(((unsigned)((s0 >> 32) & 0xFFFFull)) |
                  (((unsigned)(s1 & 0xFFFFull)) << 16)),         // s0[47:32] | s1[15:0]
            (int)(unsigned)((s1 >> 16) & 0xFFFFFFFFull)};        // s1[47:16]
  v4i g2 = {(int)td2, 0, 0, 0};                                  // tensor_dim2
  v4i g3 = {0, 0, 0, 0};
#if __has_include(<hip/amd_detail/amd_gfx1250_TDM.h>)
  v8i g4 = {0, 0, 0, 0, 0, 0, 0, 0};
  __builtin_amdgcn_tensor_load_to_lds(g0, g1, g2, g3, g4, 0);
#else
  __builtin_amdgcn_tensor_load_to_lds(g0, g1, g2, g3, 0);
#endif
}

// ---------------- misc elementwise kernels ----------------
__global__ void cvt_f32_bf16(const float* __restrict__ src, __bf16* __restrict__ dst, int n) {
  int i = blockIdx.x * blockDim.x + threadIdx.x;
  if (i < n) dst[i] = (__bf16)src[i];
}

// Router logits + softmax probs. One 64-thread block per token.
__global__ void router_kernel(const float* __restrict__ x, const float* __restrict__ rw,
                              const float* __restrict__ rb, float* __restrict__ probs) {
  __shared__ float red[4 * 64];
  const int t = blockIdx.x, tid = threadIdx.x;
  const float* xr = x + (size_t)t * DD;
  float a0 = 0.f, a1 = 0.f, a2 = 0.f, a3 = 0.f;
  for (int d = tid; d < DD; d += 64) {
    float xv = xr[d];
    a0 += xv * rw[d * 4 + 0]; a1 += xv * rw[d * 4 + 1];
    a2 += xv * rw[d * 4 + 2]; a3 += xv * rw[d * 4 + 3];
  }
  red[tid] = a0; red[64 + tid] = a1; red[128 + tid] = a2; red[192 + tid] = a3;
  __syncthreads();
  for (int s = 32; s > 0; s >>= 1) {
    if (tid < s) {
      red[tid] += red[tid + s];             red[64 + tid] += red[64 + tid + s];
      red[128 + tid] += red[128 + tid + s]; red[192 + tid] += red[192 + tid + s];
    }
    __syncthreads();
  }
  if (tid == 0) {
    float l0 = red[0] + rb[0], l1 = red[64] + rb[1], l2 = red[128] + rb[2], l3 = red[192] + rb[3];
    float m = fmaxf(fmaxf(l0, l1), fmaxf(l2, l3));
    float e0 = __expf(l0 - m), e1 = __expf(l1 - m), e2 = __expf(l2 - m), e3 = __expf(l3 - m);
    float inv = 1.0f / (e0 + e1 + e2 + e3);
    probs[t * 4 + 0] = e0 * inv; probs[t * 4 + 1] = e1 * inv;
    probs[t * 4 + 2] = e2 * inv; probs[t * 4 + 3] = e3 * inv;
  }
}

// Expert-preferred greedy routing via in-LDS bitonic sort per batch.
__global__ void route_assign(const float* __restrict__ probs, int* __restrict__ assigned,
                             int* __restrict__ dtok, float* __restrict__ rp,
                             float* __restrict__ outA, float* __restrict__ outRp) {
  __shared__ float ss[NN];
  __shared__ int   si[NN];
  __shared__ int   asg[NN];
  const int b = blockIdx.x, tid = threadIdx.x, base = b * NN;
  const int caps[4] = {1024, 512, 256, 256};
  for (int i = tid; i < NN; i += 256) asg[i] = -1;
  __syncthreads();
  for (int e = 3; e >= 0; --e) {
    for (int i = tid; i < NN; i += 256) {
      ss[i] = (asg[i] < 0) ? probs[(size_t)(base + i) * 4 + e] : -1e30f;
      si[i] = i;
    }
    __syncthreads();
    for (int k = 2; k <= NN; k <<= 1) {
      for (int j = k >> 1; j > 0; j >>= 1) {
        for (int i = tid; i < NN; i += 256) {
          int x2 = i ^ j;
          if (x2 > i) {
            float a = ss[i], c = ss[x2]; int ia = si[i], ic = si[x2];
            bool before = (a > c) || (a == c && ia < ic);   // desc by score, ties by idx
            bool dirUp = ((i & k) == 0);
            if (dirUp ? !before : before) { ss[i] = c; ss[x2] = a; si[i] = ic; si[x2] = ia; }
          }
        }
        __syncthreads();
      }
    }
    for (int i = tid; i < caps[e]; i += 256) {
      int t = si[i];
      if (asg[t] < 0) asg[t] = e;
    }
    __syncthreads();
  }
  for (int i = tid; i < NN; i += 256) {
    int a = asg[i]; if (a < 0) a = 0;
    float p = probs[(size_t)(base + i) * 4 + a];
    assigned[base + i] = a;
    dtok[base + i] = 64 << a;
    rp[base + i] = p;
    outA[base + i] = (float)a;
    outRp[base + i] = p;
  }
}

// LayerNorm over full D, then nested feature mask, store bf16.
__global__ void ln_mask_kernel(const float* __restrict__ x, const float* __restrict__ g,
                               const float* __restrict__ bt, const int* __restrict__ dtok,
                               __bf16* __restrict__ out) {
  __shared__ float red[256];
  const int t = blockIdx.x, tid = threadIdx.x;
  const float* xr = x + (size_t)t * DD;
  float v0 = xr[tid], v1 = xr[tid + 256];
  red[tid] = v0 + v1; __syncthreads();
  for (int s = 128; s > 0; s >>= 1) { if (tid < s) red[tid] += red[tid + s]; __syncthreads(); }
  float mu = red[0] * (1.0f / DD);
  __syncthreads();
  float d0 = v0 - mu, d1 = v1 - mu;
  red[tid] = d0 * d0 + d1 * d1; __syncthreads();
  for (int s = 128; s > 0; s >>= 1) { if (tid < s) red[tid] += red[tid + s]; __syncthreads(); }
  float inv = rsqrtf(red[0] * (1.0f / DD) + 1e-5f);
  int dt = dtok[t];
  __bf16* orow = out + (size_t)t * DD;
  float h0 = d0 * inv * g[tid] + bt[tid];
  float h1 = d1 * inv * g[tid + 256] + bt[tid + 256];
  orow[tid]       = (__bf16)((tid < dt) ? h0 : 0.0f);
  orow[tid + 256] = (__bf16)((tid + 256 < dt) ? h1 : 0.0f);
}

// ---------------- generic WMMA GEMM: C = A[M,K](bf16) x B[K,N](bf16) ----------------
// Block tile 64x128, 8 waves of 32x32, K-tile 64, TDM-staged double-buffered LDS.
// EPI 0: QKV  (mask per D-group, fold 1/sqrt(HD) into q) -> bf16
// EPI 1: W_o  (+bias, mask, + x residual)                 -> f32 (z)
// EPI 2: W1   (+bias, hidden mask, tanh-GELU)             -> bf16 (hid)
// EPI 3: W2   (+bias, mask, z + (alpha*rp+1)*zp)          -> f32 (d_out)
template <int EPI>
__global__ __launch_bounds__(256) void gemm_wmma(
    const __bf16* __restrict__ A, const __bf16* __restrict__ Bw,
    int M, int K, int Nc,
    const int* __restrict__ dtok, const float* __restrict__ bias,
    const float* __restrict__ resid, const float* __restrict__ rp,
    const float* __restrict__ alpha,
    __bf16* __restrict__ outH, float* __restrict__ outF) {
  __shared__ __attribute__((aligned(16))) __bf16 As[2][64 * 64];   // [buf][row][k]
  __shared__ __attribute__((aligned(16))) __bf16 Bs[2][128 * 64];  // [buf][n][k] (transposed)
  (void)M;
  const int tid = threadIdx.x;
  const int row0 = blockIdx.y * 64;
  const int n0   = blockIdx.x * 128;
  const int wid = tid >> 5, lane = tid & 31, hs = lane >> 4, r = lane & 15;
  const int wm = (wid >> 2) * 32, wn = (wid & 3) * 32;
  const unsigned ldsA = (unsigned)(uintptr_t)(void*)&As[0][0];
  const unsigned ldsB = (unsigned)(uintptr_t)(void*)&Bs[0][0];
  v8f zv = {0, 0, 0, 0, 0, 0, 0, 0};
  v8f acc[2][2] = {{zv, zv}, {zv, zv}};
  // software pipeline: prime buffer 0
  if (wid == 0) {
    tdm_load(ldsA, &A[(size_t)row0 * K], 64, 64, 0, 64, 64, 0, (unsigned long long)K, 0ull);
    tdm_load(ldsB, &Bw[(size_t)n0],      1, 64, 128, 1, 64, 128, (unsigned long long)Nc, 1ull);
  }
  int ib = 0;
  for (int k0 = 0; k0 < K; k0 += 64, ib ^= 1) {
    if (wid == 0) {
      if (k0 + 64 < K) {
        const int nb = ib ^ 1, k1 = k0 + 64;
        tdm_load(ldsA + nb * (64 * 64 * 2), &A[(size_t)row0 * K + k1],
                 64, 64, 0, 64, 64, 0, (unsigned long long)K, 0ull);
        tdm_load(ldsB + nb * (128 * 64 * 2), &Bw[(size_t)k1 * Nc + n0],
                 1, 64, 128, 1, 64, 128, (unsigned long long)Nc, 1ull);
        __builtin_amdgcn_s_wait_tensorcnt(2);   // current pair done, next pair in flight
      } else {
        __builtin_amdgcn_s_wait_tensorcnt(0);
      }
    }
    __syncthreads();   // current tiles ready for all waves
    const __bf16* Ab = &As[ib][0];
    const __bf16* Bb = &Bs[ib][0];
#pragma unroll
    for (int kc = 0; kc < 64; kc += 32) {
      v16bf a0 = a_frag32((const unsigned*)&Ab[(wm + r) * 64 + kc], hs);
      v16bf a1 = a_frag32((const unsigned*)&Ab[(wm + 16 + r) * 64 + kc], hs);
      v16bf b0 = b_frag32((const unsigned*)&Bb[(wn + r) * 64 + kc], hs);
      v16bf b1 = b_frag32((const unsigned*)&Bb[(wn + 16 + r) * 64 + kc], hs);
      acc[0][0] = wmma_bf16(a0, b0, acc[0][0]);
      acc[0][1] = wmma_bf16(a0, b1, acc[0][1]);
      acc[1][0] = wmma_bf16(a1, b0, acc[1][0]);
      acc[1][1] = wmma_bf16(a1, b1, acc[1][1]);
    }
    __syncthreads();   // all waves done reading buf[ib] before TDM retargets it
  }
#pragma unroll
  for (int i = 0; i < 2; ++i) {
#pragma unroll
    for (int j = 0; j < 2; ++j) {
      const int col = n0 + wn + j * 16 + r;
#pragma unroll
      for (int v = 0; v < 8; ++v) {
        const int row = row0 + wm + i * 16 + v + hs * 8;
        float f = acc[i][j][v];
        const int dt = dtok[row];
        if (EPI == 0) {
          const int cm = col & (DD - 1);
          const float qs = (col < DD) ? 0.125f : 1.0f;   // 1/sqrt(64) folded into q
          f = (cm < dt) ? f * qs : 0.0f;
          outH[(size_t)row * D3 + col] = (__bf16)f;
        } else if (EPI == 1) {
          f += bias[col];
          f = (col < dt) ? f : 0.0f;
          outF[(size_t)row * DD + col] = resid[(size_t)row * DD + col] + f;
        } else if (EPI == 2) {
          f += bias[col];
          f = (col < 4 * dt) ? f : 0.0f;
          float g = 0.5f * f * (1.0f + tanhf(0.7978845608028654f * (f + 0.044715f * f * f * f)));
          outH[(size_t)row * RD + col] = (__bf16)g;
        } else {
          f += bias[col];
          f = (col < dt) ? f : 0.0f;
          float gate = alpha[0] * rp[row] + 1.0f;
          outF[(size_t)row * DD + col] = resid[(size_t)row * DD + col] + gate * f;
        }
      }
    }
  }
}

// ---------------- flash attention: one (b, h, 128 q-rows) per block ----------------
__global__ __launch_bounds__(256) void attn_kernel(const __bf16* __restrict__ qkv,
                                                   const int* __restrict__ dtok,
                                                   __bf16* __restrict__ o) {
  __shared__ __attribute__((aligned(16))) __bf16 Ks[2][32 * 64];    // [buf][key][hd]
  __shared__ __attribute__((aligned(16))) __bf16 Vt[2][64 * 32];    // [buf][hd][key]
  __shared__ __attribute__((aligned(16))) __bf16 Ps[8][16 * 32];    // per-wave P tiles
  const int tid = threadIdx.x;
  const int qc = blockIdx.x, h = blockIdx.y, b = blockIdx.z;
  const int wid = tid >> 5, lane = tid & 31, hs = lane >> 4, r = lane & 15;
  const int q0 = qc * 128 + wid * 16;
  const unsigned* qg = (const unsigned*)qkv;
  const unsigned ldsK = (unsigned)(uintptr_t)(void*)&Ks[0][0];
  const unsigned ldsV = (unsigned)(uintptr_t)(void*)&Vt[0][0];
  // Q fragments for this wave's 16 rows (q already scaled + masked at QKV epilogue)
  const size_t qrow = ((size_t)(b * NN + q0 + r) * D3 + h * HDIM) >> 1;
  v16bf aq0 = a_frag32(qg + qrow, hs);
  v16bf aq1 = a_frag32(qg + qrow + 16, hs);
  v8f zv = {0, 0, 0, 0, 0, 0, 0, 0};
  v8f oacc[4] = {zv, zv, zv, zv};
  float mrun[8], lrun[8];
#pragma unroll
  for (int i = 0; i < 8; ++i) { mrun[i] = -1e30f; lrun[i] = 0.0f; }
  // prime buffer 0 with keys/values [0,32)
  if (wid == 0) {
    tdm_load(ldsK, &qkv[(size_t)(b * NN) * D3 + DD + h * HDIM],
             64, 32, 0, 64, 32, 0, (unsigned long long)D3, 0ull);
    tdm_load(ldsV, &qkv[(size_t)(b * NN) * D3 + 2 * DD + h * HDIM],
             1, 32, 64, 1, 32, 64, (unsigned long long)D3, 1ull);
  }
  int ib = 0;
  for (int j0 = 0; j0 < NN; j0 += 32, ib ^= 1) {
    if (wid == 0) {
      if (j0 + 32 < NN) {
        const int nb = ib ^ 1, j1 = j0 + 32;
        tdm_load(ldsK + nb * (32 * 64 * 2), &qkv[(size_t)(b * NN + j1) * D3 + DD + h * HDIM],
                 64, 32, 0, 64, 32, 0, (unsigned long long)D3, 0ull);
        tdm_load(ldsV + nb * (64 * 32 * 2), &qkv[(size_t)(b * NN + j1) * D3 + 2 * DD + h * HDIM],
                 1, 32, 64, 1, 32, 64, (unsigned long long)D3, 1ull);
        __builtin_amdgcn_s_wait_tensorcnt(2);   // current K/V pair done
      } else {
        __builtin_amdgcn_s_wait_tensorcnt(0);
      }
    }
    __syncthreads();
    const __bf16* Kb = &Ks[ib][0];
    const __bf16* Vb = &Vt[ib][0];
    v8f s[2];
#pragma unroll
    for (int nb2 = 0; nb2 < 2; ++nb2) {
      v8f sv = zv;
      const unsigned* krow = (const unsigned*)&Kb[(nb2 * 16 + r) * 64];
      sv = wmma_bf16(aq0, b_frag32(krow, hs), sv);        // hd 0..31
      sv = wmma_bf16(aq1, b_frag32(krow + 16, hs), sv);   // hd 32..63
      s[nb2] = sv;
    }
    float p0[8], p1[8];
#pragma unroll
    for (int i = 0; i < 8; ++i) {
      float mc = rmax16(fmaxf(s[0][i], s[1][i]));
      float mn = fmaxf(mrun[i], mc);
      float sc = __expf(mrun[i] - mn);
      mrun[i] = mn;
      float e0 = __expf(s[0][i] - mn);
      float e1 = __expf(s[1][i] - mn);
      p0[i] = e0; p1[i] = e1;
      lrun[i] = lrun[i] * sc + rsum16(e0 + e1);
#pragma unroll
      for (int cb = 0; cb < 4; ++cb) oacc[cb][i] *= sc;
    }
    // P (D-layout regs) -> wave-private LDS -> A-layout fragment
    __bf16* pw = &Ps[wid][0];
#pragma unroll
    for (int i = 0; i < 8; ++i) {
      pw[(i + hs * 8) * 32 + r]      = (__bf16)p0[i];
      pw[(i + hs * 8) * 32 + 16 + r] = (__bf16)p1[i];
    }
    v16bf pf = a_frag32((const unsigned*)&pw[r * 32], hs);
#pragma unroll
    for (int cb = 0; cb < 4; ++cb) {
      oacc[cb] = wmma_bf16(pf, b_frag32((const unsigned*)&Vb[(cb * 16 + r) * 32], hs), oacc[cb]);
    }
    __syncthreads();   // all waves done with buf[ib] before TDM retargets it
  }
#pragma unroll
  for (int cb = 0; cb < 4; ++cb) {
    const int col = h * HDIM + cb * 16 + r;
#pragma unroll
    for (int i = 0; i < 8; ++i) {
      const int tok = q0 + i + hs * 8;
      const int dt = dtok[b * NN + tok];
      float val = oacc[cb][i] / lrun[i];
      o[(size_t)(b * NN + tok) * DD + col] = (__bf16)((col < dt) ? val : 0.0f);
    }
  }
}

// ---------------- launcher ----------------
extern "C" void kernel_launch(void* const* d_in, const int* in_sizes, int n_in,
                              void* d_out, int out_size, void* d_ws, size_t ws_size,
                              hipStream_t stream) {
  (void)in_sizes; (void)n_in; (void)out_size; (void)ws_size;
  const float* x     = (const float*)d_in[0];
  const float* r_w   = (const float*)d_in[1];
  const float* r_b   = (const float*)d_in[2];
  const float* g1    = (const float*)d_in[3];
  const float* b1    = (const float*)d_in[4];
  const float* g2    = (const float*)d_in[5];
  const float* b2    = (const float*)d_in[6];
  const float* w_qkv = (const float*)d_in[7];
  const float* w_o   = (const float*)d_in[8];
  const float* b_o   = (const float*)d_in[9];
  const float* w1    = (const float*)d_in[10];
  const float* b1f   = (const float*)d_in[11];
  const float* w2    = (const float*)d_in[12];
  const float* b2f   = (const float*)d_in[13];
  const float* alpha = (const float*)d_in[14];

  float* outF = (float*)d_out;                    // [B,N,D] f32
  float* outA = outF + (size_t)MTOT * DD;         // assigned (as float)
  float* outR = outA + MTOT;                      // rp

  size_t off = 0;
  char* wsb = (char*)d_ws;
  auto carve = [&](size_t bytes) -> char* {
    char* p = wsb + off;
    off += (bytes + 255) & ~(size_t)255;
    return p;
  };
  float*  probs    = (float*)carve((size_t)MTOT * 4 * sizeof(float));
  int*    assigned = (int*)  carve((size_t)MTOT * sizeof(int));
  int*    dtok     = (int*)  carve((size_t)MTOT * sizeof(int));
  float*  rp       = (float*)carve((size_t)MTOT * sizeof(float));
  __bf16* wqkvh    = (__bf16*)carve((size_t)DD * D3 * 2);
  __bf16* woh      = (__bf16*)carve((size_t)DD * DD * 2);
  __bf16* w1h      = (__bf16*)carve((size_t)DD * RD * 2);
  __bf16* w2h      = (__bf16*)carve((size_t)RD * DD * 2);
  __bf16* hln      = (__bf16*)carve((size_t)MTOT * DD * 2);
  __bf16* qkv      = (__bf16*)carve((size_t)MTOT * D3 * 2);
  __bf16* oat      = (__bf16*)carve((size_t)MTOT * DD * 2);
  float*  zres     = (float*) carve((size_t)MTOT * DD * sizeof(float));
  __bf16* hm       = (__bf16*)carve((size_t)MTOT * DD * 2);
  __bf16* hid      = (__bf16*)carve((size_t)MTOT * RD * 2);

  // 1) weights -> bf16
  {
    int n;
    n = DD * D3; cvt_f32_bf16<<<(n + 255) / 256, 256, 0, stream>>>(w_qkv, wqkvh, n);
    n = DD * DD; cvt_f32_bf16<<<(n + 255) / 256, 256, 0, stream>>>(w_o, woh, n);
    n = DD * RD; cvt_f32_bf16<<<(n + 255) / 256, 256, 0, stream>>>(w1, w1h, n);
    n = RD * DD; cvt_f32_bf16<<<(n + 255) / 256, 256, 0, stream>>>(w2, w2h, n);
  }
  // 2) router probs + greedy capacity assignment
  router_kernel<<<MTOT, 64, 0, stream>>>(x, r_w, r_b, probs);
  route_assign<<<BB, 256, 0, stream>>>(probs, assigned, dtok, rp, outA, outR);
  // 3) LN1 * m -> bf16
  ln_mask_kernel<<<MTOT, 256, 0, stream>>>(x, g1, b1, dtok, hln);
  // 4) QKV GEMM (mask + q scale)
  gemm_wmma<0><<<dim3(D3 / 128, MTOT / 64), 256, 0, stream>>>(
      hln, wqkvh, MTOT, DD, D3, dtok, nullptr, nullptr, nullptr, nullptr, qkv, nullptr);
  // 5) flash attention -> masked o (bf16)
  attn_kernel<<<dim3(NN / 128, HH, BB), 256, 0, stream>>>(qkv, dtok, oat);
  // 6) output proj + residual -> z (f32)
  gemm_wmma<1><<<dim3(DD / 128, MTOT / 64), 256, 0, stream>>>(
      oat, woh, MTOT, DD, DD, dtok, b_o, x, nullptr, nullptr, nullptr, zres);
  // 7) LN2 * m -> bf16
  ln_mask_kernel<<<MTOT, 256, 0, stream>>>(zres, g2, b2, dtok, hm);
  // 8) FFN up (+bias, hidden mask, GELU) -> hid (bf16)
  gemm_wmma<2><<<dim3(RD / 128, MTOT / 64), 256, 0, stream>>>(
      hm, w1h, MTOT, DD, RD, dtok, b1f, nullptr, nullptr, nullptr, hid, nullptr);
  // 9) FFN down (+bias, mask, gated residual) -> d_out
  gemm_wmma<3><<<dim3(DD / 128, MTOT / 64), 256, 0, stream>>>(
      hid, w2h, MTOT, RD, DD, dtok, b2f, zres, rp, alpha, nullptr, outF);
}
